// Attn_30623116820602
// MI455X (gfx1250) — matrix operand order repaired and checked by reading, not compiled, not run on hardware
//
#include <hip/hip_runtime.h>
#include <hip/hip_bf16.h>
#include <math.h>

typedef __bf16 bf16_t;
typedef __attribute__((ext_vector_type(16))) __bf16 v16bf;
typedef __attribute__((ext_vector_type(8)))  __bf16 v8bf;
typedef __attribute__((ext_vector_type(8)))  float  v8f;

#define D_MODEL 768
#define N_HEADS 6
#define HD      128
#define RANK    192
#define BATCH   128
#define SEQ     256
#define NTOK    (BATCH * SEQ)   // 32768

static __device__ __forceinline__ bf16_t f2bf(float f) { return (bf16_t)f; }

static __device__ __forceinline__ v8f v8f_zero() {
  v8f z = {0.f, 0.f, 0.f, 0.f, 0.f, 0.f, 0.f, 0.f};
  return z;
}

// A-fragment (16x32 bf16, MxK). Lane holds row m = lane&15; half = lane>>4.
// Element e in [0,7]  -> K = k0 + 16*half + e        (contiguous 16B)
// Element e in [8,15] -> K = k0 + 16*half + 8 + e-8  (contiguous 16B)
static __device__ __forceinline__ v16bf load_a_frag(const bf16_t* row_base, int half, int k0) {
  const bf16_t* p = row_base + k0 + 16 * half;
  v8bf lo = *(const v8bf*)(p);
  v8bf hi = *(const v8bf*)(p + 8);
  return __builtin_shufflevector(lo, hi, 0,1,2,3,4,5,6,7,8,9,10,11,12,13,14,15);
}

// B-fragment (32x16 bf16, KxN) from a transposed (N-major) array Bt[n][k].
// Lane holds column n = lane&15; element e -> K = k0 + 16*half + e (contiguous 32B)
static __device__ __forceinline__ v16bf load_b_frag(const bf16_t* bt_row_base, int half, int k0) {
  return *(const v16bf*)(bt_row_base + k0 + 16 * half);
}

static __device__ __forceinline__ v8f wmma_bf16(v16bf a, v16bf b, v8f c) {
  return __builtin_amdgcn_wmma_f32_16x16x32_bf16(false, a, false, b, (short)0, c, false, false);
}

// 32xN GEMM core, K-loop FULLY UNROLLED: straight-line SSA code lets the
// register allocator coalesce load quads directly into WMMA operand groups
// (no rotation movs, no hazard nops) and the scheduler hoists the s_clause'd
// loads across preceding WMMAs for latency hiding.
template <int KDIM, int NT>
static __device__ __forceinline__ void gemm_full(
    const bf16_t* __restrict__ arow0, const bf16_t* __restrict__ arow1,
    const bf16_t* __restrict__ btrow, int half, v8f* acc0, v8f* acc1) {
#pragma unroll
  for (int k0 = 0; k0 < KDIM; k0 += 32) {
    v16bf a0 = load_a_frag(arow0, half, k0);
    v16bf a1 = load_a_frag(arow1, half, k0);
#pragma unroll
    for (int j = 0; j < NT; ++j) {
      v16bf bf = load_b_frag(btrow + (size_t)j * 16 * KDIM, half, k0);
      acc0[j] = wmma_bf16(a0, bf, acc0[j]);
      acc1[j] = wmma_bf16(a1, bf, acc1[j]);
    }
  }
}

// ---------------- elementwise conversion / transpose ----------------

__global__ void k_f32_to_bf16(const float* __restrict__ src, bf16_t* __restrict__ dst, int n) {
  int i = blockIdx.x * blockDim.x + threadIdx.x;
  if (i < n) dst[i] = f2bf(src[i]);
}

// src fp32 [R, C] row-major  ->  dst bf16 [C, R] row-major (i.e. transposed)
__global__ void k_transpose_to_bf16(const float* __restrict__ src, bf16_t* __restrict__ dst,
                                    int R, int C) {
  int i = blockIdx.x * blockDim.x + threadIdx.x;
  if (i < R * C) {
    int r = i / C, c = i - r * C;
    dst[(size_t)c * R + r] = f2bf(src[i]);
  }
}

// ---------------- GEMM1: X[NTOK,768] @ A[768,192] -> XA[NTOK,192] (bf16) ----------------
// 32x64 wave tile; 4 waves/block in M; grid (256, 3, 3)

__global__ void __launch_bounds__(128)
k_gemm_rank(const bf16_t* __restrict__ X,
            const bf16_t* __restrict__ At0, const bf16_t* __restrict__ At1,
            const bf16_t* __restrict__ At2,
            bf16_t* __restrict__ Y0, bf16_t* __restrict__ Y1, bf16_t* __restrict__ Y2) {
  const int lane = threadIdx.x & 31;
  const int wave = threadIdx.x >> 5;
  const int n = lane & 15, half = lane >> 4;
  const int m0 = (blockIdx.x * 4 + wave) * 32;
  const int ncol0 = blockIdx.y * 64;
  const int z = blockIdx.z;
  const bf16_t* At = (z == 0) ? At0 : (z == 1) ? At1 : At2;   // [RANK, 768]
  bf16_t* Y = (z == 0) ? Y0 : (z == 1) ? Y1 : Y2;

  const bf16_t* arow0 = X + (size_t)(m0 + n) * D_MODEL;
  const bf16_t* arow1 = X + (size_t)(m0 + 16 + n) * D_MODEL;
  const bf16_t* btrow = At + (size_t)(ncol0 + n) * D_MODEL;

  v8f acc0[4], acc1[4];
#pragma unroll
  for (int j = 0; j < 4; ++j) { acc0[j] = v8f_zero(); acc1[j] = v8f_zero(); }

  gemm_full<D_MODEL, 4>(arow0, arow1, btrow, half, acc0, acc1);

#pragma unroll
  for (int j = 0; j < 4; ++j)
#pragma unroll
    for (int r = 0; r < 8; ++r) {
      int rr = r + 8 * half;
      Y[(size_t)(m0 + rr) * RANK + ncol0 + j * 16 + n]      = f2bf(acc0[j][r]);
      Y[(size_t)(m0 + 16 + rr) * RANK + ncol0 + j * 16 + n] = f2bf(acc1[j][r]);
    }
}

// ---------------- GEMM2 + RoPE: XA[NTOK,192] @ B[192,768] -> Q/K [B,H,T,HD], V^T [B,H,HD,T] ----
// wave computes 32 rows x one full head (128 cols) so RoPE pairs (d, d+64) are
// fragment nt and nt+4 in the SAME lane. K-loop (6 steps) fully unrolled.
// grid (256, 6, 3)

__global__ void __launch_bounds__(128)
k_gemm_proj(const bf16_t* __restrict__ XA0, const bf16_t* __restrict__ XA1,
            const bf16_t* __restrict__ XA2,
            const bf16_t* __restrict__ Bt0, const bf16_t* __restrict__ Bt1,
            const bf16_t* __restrict__ Bt2,
            bf16_t* __restrict__ Qo, bf16_t* __restrict__ Ko, bf16_t* __restrict__ Vt) {
  const int lane = threadIdx.x & 31;
  const int wave = threadIdx.x >> 5;
  const int n = lane & 15, half = lane >> 4;
  const int m0 = (blockIdx.x * 4 + wave) * 32;
  const int h = blockIdx.y;
  const int z = blockIdx.z;
  const bf16_t* XA = (z == 0) ? XA0 : (z == 1) ? XA1 : XA2;   // [NTOK, RANK]
  const bf16_t* Bt = (z == 0) ? Bt0 : (z == 1) ? Bt1 : Bt2;   // [768, RANK]

  const bf16_t* arow0 = XA + (size_t)(m0 + n) * RANK;
  const bf16_t* arow1 = XA + (size_t)(m0 + 16 + n) * RANK;
  const bf16_t* btrow = Bt + (size_t)(h * HD + n) * RANK;

  v8f acc0[8], acc1[8];
#pragma unroll
  for (int j = 0; j < 8; ++j) { acc0[j] = v8f_zero(); acc1[j] = v8f_zero(); }

  gemm_full<RANK, 8>(arow0, arow1, btrow, half, acc0, acc1);

  if (z < 2) {
    bf16_t* dst = (z == 0) ? Qo : Ko;
#pragma unroll
    for (int nt = 0; nt < 4; ++nt) {
      int d = nt * 16 + n;  // 0..63
      float inv = __powf(10000.0f, -(float)d * (1.0f / 64.0f));
#pragma unroll
      for (int mt = 0; mt < 2; ++mt)
#pragma unroll
        for (int r = 0; r < 8; ++r) {
          int row = m0 + mt * 16 + r + 8 * half;
          int b = row >> 8, t = row & 255;
          float ang = (float)t * inv;
          float cs = __cosf(ang), sn = __sinf(ang);
          float x1 = mt ? acc1[nt][r]     : acc0[nt][r];      // q[d], d < 64
          float x2 = mt ? acc1[nt + 4][r] : acc0[nt + 4][r];  // q[d+64]
          size_t base = ((size_t)(b * N_HEADS + h) * SEQ + t) * HD;
          dst[base + d]      = f2bf(x1 * cs - x2 * sn);  // rot = -q[d+64]
          dst[base + d + 64] = f2bf(x2 * cs + x1 * sn);  // rot =  q[d]
        }
    }
  } else {
#pragma unroll
    for (int nt = 0; nt < 8; ++nt) {
      int d = nt * 16 + n;
#pragma unroll
      for (int mt = 0; mt < 2; ++mt)
#pragma unroll
        for (int r = 0; r < 8; ++r) {
          int row = m0 + mt * 16 + r + 8 * half;
          int b = row >> 8, t = row & 255;
          float v = mt ? acc1[nt][r] : acc0[nt][r];
          Vt[((size_t)(b * N_HEADS + h) * HD + d) * SEQ + t] = f2bf(v);
        }
    }
  }
}

// ---------------- flash attention: one wave per 16-row q-tile ----------------
// grid (16 q-tiles, 6 heads, 128 batch), block = 32 (one wave)

__global__ void __launch_bounds__(32)
k_attn(const bf16_t* __restrict__ Q, const bf16_t* __restrict__ K,
       const bf16_t* __restrict__ Vt, bf16_t* __restrict__ O) {
  __shared__ __attribute__((aligned(32))) bf16_t lds_p[16 * 32];
  const int lane = threadIdx.x & 31;
  const int n = lane & 15, half = lane >> 4;
  const int qt = blockIdx.x;
  const int h  = blockIdx.y;
  const int b  = blockIdx.z;

  const bf16_t* Qp = Q  + ((size_t)(b * N_HEADS + h) * SEQ) * HD;  // [T,128]
  const bf16_t* Kp = K  + ((size_t)(b * N_HEADS + h) * SEQ) * HD;  // [T,128]
  const bf16_t* Vp = Vt + ((size_t)(b * N_HEADS + h) * HD) * SEQ;  // [128,T]

  v16bf qf[4];
#pragma unroll
  for (int j = 0; j < 4; ++j)
    qf[j] = load_a_frag(Qp + (size_t)(qt * 16 + n) * HD, half, j * 32);

  v8f o[8];
#pragma unroll
  for (int j = 0; j < 8; ++j) o[j] = v8f_zero();
  float mrow[8], lrow[8];
#pragma unroll
  for (int r = 0; r < 8; ++r) { mrow[r] = -__builtin_inff(); lrow[r] = 0.0f; }

  const float scale = 0.08838834764831845f;  // 1/sqrt(128)
  const int nkb = (qt * 16 + 15) / 32 + 1;   // 32-key blocks under causal limit

  for (int kb = 0; kb < nkb; ++kb) {
    if (kb + 1 < nkb) {
      // hint the next K block toward the caches (global_prefetch_b8)
      __builtin_prefetch(Kp + (size_t)((kb + 1) * 32 + lane) * HD, 0, 1);
    }
    v8f s0 = v8f_zero(), s1 = v8f_zero();
#pragma unroll
    for (int j = 0; j < 4; ++j) {
      v16bf kf0 = load_b_frag(Kp + (size_t)(kb * 32 + n) * HD, half, j * 32);
      v16bf kf1 = load_b_frag(Kp + (size_t)(kb * 32 + 16 + n) * HD, half, j * 32);
      s0 = wmma_bf16(qf[j], kf0, s0);
      s1 = wmma_bf16(qf[j], kf1, s1);
    }

    __syncthreads();  // WAR on lds_p vs previous iteration's fragment reads
#pragma unroll
    for (int r = 0; r < 8; ++r) {
      int row = qt * 16 + r + 8 * half;
      int k0i = kb * 32 + n;
      int k1i = kb * 32 + 16 + n;
      float e0 = (k0i <= row) ? s0[r] * scale : -__builtin_inff();
      float e1 = (k1i <= row) ? s1[r] * scale : -__builtin_inff();
      float rm = fmaxf(e0, e1);
#pragma unroll
      for (int d = 1; d < 16; d <<= 1) rm = fmaxf(rm, __shfl_xor(rm, d, 16));
      float newm  = fmaxf(mrow[r], rm);
      float alpha = __expf(mrow[r] - newm);   // first iter: exp(-inf)=0
      float p0 = __expf(e0 - newm);
      float p1 = __expf(e1 - newm);
      float rs = p0 + p1;
#pragma unroll
      for (int d = 1; d < 16; d <<= 1) rs += __shfl_xor(rs, d, 16);
      lrow[r] = lrow[r] * alpha + rs;
      mrow[r] = newm;
#pragma unroll
      for (int nt = 0; nt < 8; ++nt) o[nt][r] *= alpha;
      lds_p[(r + 8 * half) * 32 + n]      = f2bf(p0);
      lds_p[(r + 8 * half) * 32 + 16 + n] = f2bf(p1);
    }
    __syncthreads();

    v16bf pf = load_a_frag(lds_p + n * 32, half, 0);   // P as 16x32 A-fragment
#pragma unroll
    for (int nt = 0; nt < 8; ++nt) {
      v16bf vf = load_b_frag(Vp + (size_t)(nt * 16 + n) * SEQ, half, kb * 32);
      o[nt] = wmma_bf16(pf, vf, o[nt]);
    }
  }

#pragma unroll
  for (int r = 0; r < 8; ++r) {
    float invl = 1.0f / lrow[r];
    int t = qt * 16 + r + 8 * half;
    size_t base = ((size_t)(b * SEQ + t)) * D_MODEL + h * HD;
#pragma unroll
    for (int nt = 0; nt < 8; ++nt)
      O[base + nt * 16 + n] = f2bf(o[nt][r] * invl);
  }
}

// ---------------- GEMM3: O[NTOK,768] @ o_w[768,768] -> out fp32 ----------------
// 32x64 wave tile; grid (256, 12)

__global__ void __launch_bounds__(128)
k_gemm_out(const bf16_t* __restrict__ A, const bf16_t* __restrict__ Bt,
           float* __restrict__ C) {
  const int lane = threadIdx.x & 31;
  const int wave = threadIdx.x >> 5;
  const int n = lane & 15, half = lane >> 4;
  const int m0 = (blockIdx.x * 4 + wave) * 32;
  const int ncol0 = blockIdx.y * 64;

  const bf16_t* arow0 = A + (size_t)(m0 + n) * D_MODEL;
  const bf16_t* arow1 = A + (size_t)(m0 + 16 + n) * D_MODEL;
  const bf16_t* btrow = Bt + (size_t)(ncol0 + n) * D_MODEL;

  v8f acc0[4], acc1[4];
#pragma unroll
  for (int j = 0; j < 4; ++j) { acc0[j] = v8f_zero(); acc1[j] = v8f_zero(); }

  gemm_full<D_MODEL, 4>(arow0, arow1, btrow, half, acc0, acc1);

#pragma unroll
  for (int j = 0; j < 4; ++j)
#pragma unroll
    for (int r = 0; r < 8; ++r) {
      int rr = r + 8 * half;
      C[(size_t)(m0 + rr) * D_MODEL + ncol0 + j * 16 + n]      = acc0[j][r];
      C[(size_t)(m0 + 16 + rr) * D_MODEL + ncol0 + j * 16 + n] = acc1[j][r];
    }
}

// ---------------- host launch ----------------

extern "C" void kernel_launch(void* const* d_in, const int* in_sizes, int n_in,
                              void* d_out, int out_size, void* d_ws, size_t ws_size,
                              hipStream_t stream) {
  (void)in_sizes; (void)n_in; (void)out_size; (void)ws_size;
  const float* x  = (const float*)d_in[0];
  const float* qA = (const float*)d_in[1];
  const float* qB = (const float*)d_in[2];
  const float* kA = (const float*)d_in[3];
  const float* kB = (const float*)d_in[4];
  const float* vA = (const float*)d_in[5];
  const float* vB = (const float*)d_in[6];
  const float* ow = (const float*)d_in[7];

  char* w = (char*)d_ws;
  size_t off = 0;
  auto alloc = [&](size_t bytes) -> bf16_t* {
    bf16_t* p = (bf16_t*)(w + off);
    off += (bytes + 255) & ~(size_t)255;
    return p;
  };
  bf16_t* Xbf = alloc((size_t)NTOK * D_MODEL * 2);
  bf16_t* qAt = alloc((size_t)RANK * D_MODEL * 2);   // qA^T  [192,768]
  bf16_t* kAt = alloc((size_t)RANK * D_MODEL * 2);
  bf16_t* vAt = alloc((size_t)RANK * D_MODEL * 2);
  bf16_t* qBt = alloc((size_t)D_MODEL * RANK * 2);   // qB^T  [768,192]
  bf16_t* kBt = alloc((size_t)D_MODEL * RANK * 2);
  bf16_t* vBt = alloc((size_t)D_MODEL * RANK * 2);
  bf16_t* owt = alloc((size_t)D_MODEL * D_MODEL * 2);
  bf16_t* XAq = alloc((size_t)NTOK * RANK * 2);
  bf16_t* XAk = alloc((size_t)NTOK * RANK * 2);
  bf16_t* XAv = alloc((size_t)NTOK * RANK * 2);
  bf16_t* Qbh = alloc((size_t)NTOK * D_MODEL * 2);   // [B,H,T,HD]
  bf16_t* Kbh = alloc((size_t)NTOK * D_MODEL * 2);   // [B,H,T,HD]
  bf16_t* Vtw = alloc((size_t)NTOK * D_MODEL * 2);   // [B,H,HD,T]
  bf16_t* Obf = alloc((size_t)NTOK * D_MODEL * 2);   // [NTOK,768]

  const int ne = NTOK * D_MODEL;
  k_f32_to_bf16<<<dim3((ne + 255) / 256), dim3(256), 0, stream>>>(x, Xbf, ne);

  const int nA = D_MODEL * RANK;
  k_transpose_to_bf16<<<dim3((nA + 255) / 256), dim3(256), 0, stream>>>(qA, qAt, D_MODEL, RANK);
  k_transpose_to_bf16<<<dim3((nA + 255) / 256), dim3(256), 0, stream>>>(kA, kAt, D_MODEL, RANK);
  k_transpose_to_bf16<<<dim3((nA + 255) / 256), dim3(256), 0, stream>>>(vA, vAt, D_MODEL, RANK);
  k_transpose_to_bf16<<<dim3((nA + 255) / 256), dim3(256), 0, stream>>>(qB, qBt, RANK, D_MODEL);
  k_transpose_to_bf16<<<dim3((nA + 255) / 256), dim3(256), 0, stream>>>(kB, kBt, RANK, D_MODEL);
  k_transpose_to_bf16<<<dim3((nA + 255) / 256), dim3(256), 0, stream>>>(vB, vBt, RANK, D_MODEL);
  const int nW = D_MODEL * D_MODEL;
  k_transpose_to_bf16<<<dim3((nW + 255) / 256), dim3(256), 0, stream>>>(ow, owt, D_MODEL, D_MODEL);

  k_gemm_rank<<<dim3(NTOK / 128, RANK / 64, 3), dim3(128), 0, stream>>>(
      Xbf, qAt, kAt, vAt, XAq, XAk, XAv);

  k_gemm_proj<<<dim3(NTOK / 128, N_HEADS, 3), dim3(128), 0, stream>>>(
      XAq, XAk, XAv, qBt, kBt, vBt, Qbh, Kbh, Vtw);

  k_attn<<<dim3(SEQ / 16, N_HEADS, BATCH), dim3(32), 0, stream>>>(Qbh, Kbh, Vtw, Obf);

  k_gemm_out<<<dim3(NTOK / 128, D_MODEL / 64), dim3(128), 0, stream>>>(
      Obf, owt, (float*)d_out);
}
